// SharedNet_clstm_57045755626075
// MI455X (gfx1250) — matrix operand order repaired and checked by reading, not compiled, not run on hardware
//
#include <hip/hip_runtime.h>
#include <hip/hip_bf16.h>

typedef __bf16 v16bf __attribute__((ext_vector_type(16)));
typedef float  v8f   __attribute__((ext_vector_type(8)));
typedef unsigned short us8 __attribute__((ext_vector_type(8)));

#define BB 16
#define TT 32
#define HH 32
#define WW 32

__device__ __forceinline__ unsigned short f2bf_bits(float f) {
  union { float f; unsigned u; } v; v.f = f;
  unsigned r = v.u + 0x7FFFu + ((v.u >> 16) & 1u);
  return (unsigned short)(r >> 16);
}

// ---------------------------------------------------------------------------
// Prepack conv weights (fp32 [M][C][3][3] == [M][K]) into WMMA A-fragment
// order (bf16): apack[((mt*KT + kt)*32 + lane)*16 + i]
// 16-bit A 16x32 layout: lane half selects K phase, VGPR pair packing.
// ---------------------------------------------------------------------------
__global__ void prepack_kernel(const float* __restrict__ w,
                               unsigned short* __restrict__ apack,
                               int M, int Kreal, int KT, int MT) {
  int idx = blockIdx.x * blockDim.x + threadIdx.x;
  int total = MT * KT * 512;
  if (idx >= total) return;
  int i    = idx & 15;
  int lane = (idx >> 4) & 31;
  int kt   = (idx >> 9) % KT;
  int mt   = (idx >> 9) / KT;
  int half = lane >> 4, mm = lane & 15;
  int v = i >> 1, p = i & 1;
  int koff = (v < 4) ? (2 * v + p + 8 * half) : (16 + 2 * (v - 4) + p + 8 * half);
  int m = mt * 16 + mm;
  int k = kt * 32 + koff;
  float val = (m < M && k < Kreal) ? w[(size_t)m * Kreal + k] : 0.0f;
  apack[idx] = f2bf_bits(val);
}

// x fp32 [B][T][2][32][32]  ->  act bf16 [T][B][2][32][32]
__global__ void convert_input_kernel(const float* __restrict__ x,
                                     unsigned short* __restrict__ act) {
  int idx = blockIdx.x * blockDim.x + threadIdx.x;
  if (idx >= TT * BB * 2 * 1024) return;
  int s = idx & 1023;
  int c = (idx >> 10) & 1;
  int b = (idx >> 11) & 15;
  int t = idx >> 15;
  act[idx] = f2bf_bits(x[(((size_t)b * TT + t) * 2 + c) * 1024 + s]);
}

// ---------------------------------------------------------------------------
// One ConvLSTM timestep: implicit-GEMM 3x3 conv via WMMA + LSTM gate math.
// Workgroup: 128 threads (4 waves), owns 16 consecutive positions in one row.
// B operand is materialized in LDS in exact WMMA-fragment order; the K loop
// is software-pipelined (prefetch kt+1 fragments before kt WMMAs) and free of
// per-iteration EXEC guards.
// ---------------------------------------------------------------------------
template <int CIN, int HID, bool RELU, bool OUTF32>
__global__ __launch_bounds__(128) void clstm_step_kernel(
    const unsigned short* __restrict__ xin,    // [B][CIN][H][W] bf16 (time t)
    const unsigned short* __restrict__ hprev,  // [B][HID][H][W] bf16
    unsigned short* __restrict__ hnext,        // [B][HID][H][W] bf16
    float* __restrict__ cbuf,                  // [B][HID][H][W] f32
    const unsigned short* __restrict__ apack,  // prepacked weights
    const float* __restrict__ bias,            // [4*HID]
    unsigned short* __restrict__ xout,         // [B][HID][H][W] bf16 (time t)
    float* __restrict__ outf,                  // d_out base [B][T][HID][H][W]
    int t) {
  constexpr int C     = CIN + HID;
  constexpr int KREAL = C * 9;
  constexpr int KT    = (KREAL + 31) / 32;
  constexpr int KPAD  = KT * 32;
  constexpr int CPAD  = (KPAD + 8) / 9;
  constexpr int M     = 4 * HID;
  constexpr int MT    = (M + 15) / 16;
  constexpr int MPAD  = MT * 16;
  constexpr int SLOTS = (MT + 3) / 4;
  constexpr bool FULL = (MT % 4 == 0);  // every wave has all SLOTS valid
  static_assert(FULL || SLOTS == 1, "partial-M case assumes a single slot");

  constexpr int BCOL_USH  = KT * 512;        // B tile in fragment order
  constexpr int PATCH_USH = CPAD * 54;       // halo patch [C][3][18]
  constexpr int TAB_USH   = KPAD;            // k -> patch offset table
  constexpr int GATE_USH  = MPAD * 16 * 2;   // gates (f32) in ushort units
  constexpr int OV_USH =
      (PATCH_USH + TAB_USH) > GATE_USH ? (PATCH_USH + TAB_USH) : GATE_USH;

  __shared__ __align__(32) unsigned short smem[BCOL_USH + OV_USH];
  unsigned short* bcol  = smem;                    // live: bcol build + K loop
  unsigned short* patch = smem + BCOL_USH;         // live: until bcol built
  unsigned short* ktab  = patch + PATCH_USH;       // live: until bcol built
  float* gates = (float*)(smem + BCOL_USH);        // live: after K loop

  const int tid = threadIdx.x;
  const int n0  = blockIdx.x * 16;
  const int b   = n0 >> 10;
  const int rem = n0 & 1023;
  const int y   = rem >> 5;
  const int x0  = rem & 31;  // 0 or 16

  // Phase 1a: stage halo patch of concat(x_t, h_{t-1}) into LDS (bf16 bits).
  for (int idx = tid; idx < PATCH_USH; idx += 128) {
    int c  = idx / 54;
    int r  = idx - c * 54;
    int ky = r / 18;
    int dx = r - ky * 18;
    int yy = y + ky - 1;
    int xx = x0 + dx - 1;
    unsigned short val = 0;
    if (c < C && yy >= 0 && yy < HH && xx >= 0 && xx < WW) {
      if (c < CIN)
        val = xin[((b * CIN + c) * HH + yy) * WW + xx];
      else if (t > 0)
        val = hprev[((b * HID + (c - CIN)) * HH + yy) * WW + xx];
    }
    patch[idx] = val;
  }
  // Phase 1b: k -> patch-offset table (does the /9, /3 math exactly once).
  for (int k = tid; k < KPAD; k += 128) {
    int ic = k / 9;
    int rr = k - ic * 9;
    int ky = rr / 3;
    int kx = rr - ky * 3;
    ktab[k] = (unsigned short)(ic * 54 + ky * 18 + kx);
  }
  __syncthreads();

  // Phase 2: materialize B operand in exact WMMA-fragment order.
  // bcol[((kt*32 + lane)*16 + i)] = B(k = kt*32 + (lane>>4)*8 + (i>>3)*16 + (i&7), n = lane&15)
  for (int ch = tid; ch < KT * 64; ch += 128) {
    int i0   = (ch & 1) * 8;
    int lane = (ch >> 1) & 31;
    int kt   = ch >> 6;
    int nn   = lane & 15;
    int kb   = kt * 32 + (lane >> 4) * 8 + i0 * 2;
    union { us8 v; unsigned short u[8]; } tmp;
#pragma unroll
    for (int j = 0; j < 8; ++j) tmp.u[j] = patch[ktab[kb + j] + nn];
    *(us8*)(bcol + ch * 8) = tmp.v;
  }
  __syncthreads();

  const int lane = tid & 31;
  const int wave = tid >> 5;
  const int half = lane >> 4;
  const int nn   = lane & 15;

#define LOADB(kt) (*(const v16bf*)(bcol + (((kt)*32 + lane) << 4)))
#define LOADA(s, kt) \
  (*(const v16bf*)(apack + ((size_t)(wave + 4 * (s)) * KT + (kt)) * 512 + lane * 16))

  v8f acc[SLOTS];
#pragma unroll
  for (int s = 0; s < SLOTS; ++s)
#pragma unroll
    for (int r = 0; r < 8; ++r) acc[s][r] = 0.0f;

  // Phase 3: software-pipelined K loop (no EXEC guards inside).
  if (FULL || wave < MT) {
    v16bf bcur = LOADB(0);
    v16bf acur[SLOTS];
#pragma unroll
    for (int s = 0; s < SLOTS; ++s) acur[s] = LOADA(s, 0);
    for (int kt = 0; kt < KT; ++kt) {
      const int ktn = (kt + 1 < KT) ? kt + 1 : kt;
      v16bf bnext = LOADB(ktn);
      v16bf anext[SLOTS];
#pragma unroll
      for (int s = 0; s < SLOTS; ++s) anext[s] = LOADA(s, ktn);
#pragma unroll
      for (int s = 0; s < SLOTS; ++s)
        acc[s] = __builtin_amdgcn_wmma_f32_16x16x32_bf16(
            false, acur[s], false, bcur, (short)0, acc[s], false, false);
      bcur = bnext;
#pragma unroll
      for (int s = 0; s < SLOTS; ++s) acur[s] = anext[s];
    }
  }
#undef LOADA
#undef LOADB

  // Phase 4: accumulators (+bias) -> LDS gates[M][16] (overlaps dead patch).
  if (FULL || wave < MT) {
#pragma unroll
    for (int s = 0; s < SLOTS; ++s) {
      int mt = wave + 4 * s;
#pragma unroll
      for (int r = 0; r < 8; ++r) {
        int m = mt * 16 + half * 8 + r;
        float bv = (m < M) ? bias[m] : 0.0f;
        gates[m * 16 + nn] = acc[s][r] + bv;
      }
    }
  }
  __syncthreads();

  // Phase 5: LSTM elementwise update.
  for (int idx = tid; idx < HID * 16; idx += 128) {
    int ch = idx >> 4;
    int n  = idx & 15;
    float gi = gates[ch * 16 + n];
    float gf = gates[(HID + ch) * 16 + n];
    float go = gates[(2 * HID + ch) * 16 + n];
    float gg = gates[(3 * HID + ch) * 16 + n];
    float si = 1.0f / (1.0f + __expf(-gi));
    float sf = 1.0f / (1.0f + __expf(-gf));
    float so = 1.0f / (1.0f + __expf(-go));
    int xx = x0 + n;
    int ci = ((b * HID + ch) * HH + y) * WW + xx;
    float cold = (t > 0) ? cbuf[ci] : 0.0f;
    float cnew = sf * cold + si * tanhf(gg);
    cbuf[ci] = cnew;
    float h = so * tanhf(cnew);
    hnext[ci] = f2bf_bits(h);
    float ho = RELU ? fmaxf(h, 0.0f) : h;
    if (OUTF32) {
      outf[(((size_t)b * TT + t) * HID + ch) * (HH * WW) + y * WW + xx] = ho;
    } else {
      xout[ci] = f2bf_bits(ho);
    }
  }
}

// ---------------------------------------------------------------------------
template <int CIN, int HID, bool RELU, bool OUTF32>
static void run_layer(const unsigned short* actIn, unsigned short* actOut,
                      unsigned short* hA, unsigned short* hB, float* cbuf,
                      const unsigned short* apack, const float* bias,
                      float* outf, hipStream_t stream) {
  const int nblocks = BB * HH * WW / 16;  // 1024
  for (int t = 0; t < TT; ++t) {
    const unsigned short* xin = actIn + (size_t)t * BB * CIN * HH * WW;
    unsigned short* xout =
        OUTF32 ? nullptr : actOut + (size_t)t * BB * HID * HH * WW;
    const unsigned short* hp = (t & 1) ? hB : hA;
    unsigned short* hn = (t & 1) ? hA : hB;
    clstm_step_kernel<CIN, HID, RELU, OUTF32><<<nblocks, 128, 0, stream>>>(
        xin, hp, hn, cbuf, apack, bias, xout, outf, t);
  }
}

extern "C" void kernel_launch(void* const* d_in, const int* in_sizes, int n_in,
                              void* d_out, int out_size, void* d_ws,
                              size_t ws_size, hipStream_t stream) {
  const float* x = (const float*)d_in[0];
  const float* w[6];
  const float* bsx[6];
  for (int i = 0; i < 6; ++i) {
    w[i] = (const float*)d_in[1 + i];
    bsx[i] = (const float*)d_in[7 + i];
  }

  static const int CINS[6] = {2, 8, 16, 32, 64, 16};
  static const int HIDS[6] = {8, 16, 32, 64, 16, 2};

  char* ws = (char*)d_ws;
  size_t off = 0;
  auto alloc = [&](size_t bytes) -> char* {
    char* p = ws + off;
    off = (off + bytes + 255) & ~(size_t)255;
    return p;
  };

  unsigned short* apack[6];
  int KTs[6], MTs[6], Ms[6], Ks[6];
  for (int l = 0; l < 6; ++l) {
    int C = CINS[l] + HIDS[l];
    Ks[l] = C * 9;
    KTs[l] = (Ks[l] + 31) / 32;
    Ms[l] = 4 * HIDS[l];
    MTs[l] = (Ms[l] + 15) / 16;
    apack[l] = (unsigned short*)alloc((size_t)MTs[l] * KTs[l] * 512 * 2);
  }
  unsigned short* actA = (unsigned short*)alloc((size_t)TT * BB * 64 * 1024 * 2);
  unsigned short* actB = (unsigned short*)alloc((size_t)TT * BB * 32 * 1024 * 2);
  unsigned short* hA   = (unsigned short*)alloc((size_t)BB * 64 * 1024 * 2);
  unsigned short* hB   = (unsigned short*)alloc((size_t)BB * 64 * 1024 * 2);
  float* cbuf          = (float*)alloc((size_t)BB * 64 * 1024 * 4);

  for (int l = 0; l < 6; ++l) {
    int total = MTs[l] * KTs[l] * 512;
    prepack_kernel<<<(total + 255) / 256, 256, 0, stream>>>(
        w[l], apack[l], Ms[l], Ks[l], KTs[l], MTs[l]);
  }
  {
    int total = TT * BB * 2 * 1024;
    convert_input_kernel<<<(total + 255) / 256, 256, 0, stream>>>(x, actA);
  }

  float* outf = (float*)d_out;
  run_layer<2, 8, false, false>(actA, actB, hA, hB, cbuf, apack[0], bsx[0], nullptr, stream);
  run_layer<8, 16, true, false>(actB, actA, hA, hB, cbuf, apack[1], bsx[1], nullptr, stream);
  run_layer<16, 32, false, false>(actA, actB, hA, hB, cbuf, apack[2], bsx[2], nullptr, stream);
  run_layer<32, 64, true, false>(actB, actA, hA, hB, cbuf, apack[3], bsx[3], nullptr, stream);
  run_layer<64, 16, false, false>(actA, actB, hA, hB, cbuf, apack[4], bsx[4], nullptr, stream);
  run_layer<16, 2, true, true>(actB, nullptr, hA, hB, cbuf, apack[5], bsx[5], outf, stream);
}
